// GPT_56710748176927
// MI455X (gfx1250) — matrix operand order repaired and checked by reading, not compiled, not run on hardware
//
#include <hip/hip_runtime.h>
#include <math.h>

// ---- problem constants ----
#define TT      2048
#define EMBD    2048
#define NH      16
#define NQG     4
#define HEADD   128
#define QKV_N   3072   // (16 + 2*4) * 128
#define SWIN    1024
#define CAPV    50.0f

typedef __attribute__((ext_vector_type(16))) __bf16 bf16x16;
typedef __attribute__((ext_vector_type(8)))  __bf16 bf16x8;
typedef __attribute__((ext_vector_type(8)))  float  f32x8;
typedef __attribute__((ext_vector_type(4)))  int    v4i;

#if defined(__AMDGCN__) && __has_builtin(__builtin_amdgcn_global_load_async_to_lds_b128) && __has_builtin(__builtin_amdgcn_s_wait_asynccnt)
#define HAVE_ASYNC_LDS 1
#else
#define HAVE_ASYNC_LDS 0
#endif

__device__ __forceinline__ unsigned short f2bf(float f) {
  unsigned u = __float_as_uint(f);
  u += 0x7fffu + ((u >> 16) & 1u);           // round-to-nearest-even
  return (unsigned short)(u >> 16);
}
__device__ __forceinline__ float bf2f(unsigned short h) {
  return __uint_as_float(((unsigned)h) << 16);
}

__device__ __forceinline__ f32x8 wmma_bf16(bf16x16 a, bf16x16 b, f32x8 c) {
  // D = A(16x32 bf16) x B(32x16 bf16) + C(16x16 f32)
  return __builtin_amdgcn_wmma_f32_16x16x32_bf16(false, a, false, b, (short)0, c,
                                                 false, false);
}

// A-matrix fragment (16x32 bf16) from [row][k] bf16 storage, stride in elems
// (must be multiple of 8). ISA 7.12.2: lanes 0-15 row=lane, k {0..7,16..23};
// lanes 16-31 same rows, k {8..15,24..31}.
__device__ __forceinline__ bf16x16 load_a_frag(const unsigned short* base, int stride) {
  const int lane = threadIdx.x & 31;
  const unsigned short* p = base + (lane & 15) * stride + ((lane >> 4) << 3);
  bf16x8 lo = *(const bf16x8*)(p);
  bf16x8 hi = *(const bf16x8*)(p + 16);
  return __builtin_shufflevector(lo, hi, 0,1,2,3,4,5,6,7,8,9,10,11,12,13,14,15);
}

// B-matrix fragment (32x16 bf16) from [n][k] bf16 storage, same k order as A:
// lanes 0-15 col=lane, k {0..7,16..23};
// lanes 16-31 same cols, k {8..15,24..31}.
__device__ __forceinline__ bf16x16 load_b_frag(const unsigned short* base, int stride) {
  const int lane = threadIdx.x & 31;
  const unsigned short* p = base + (lane & 15) * stride + ((lane >> 4) << 3);
  bf16x8 lo = *(const bf16x8*)(p);
  bf16x8 hi = *(const bf16x8*)(p + 16);
  return __builtin_shufflevector(lo, hi, 0,1,2,3,4,5,6,7,8,9,10,11,12,13,14,15);
}

// =====================================================================
// C[M,N] = A[M,K] @ B[N,K]^T  (all f32 in memory), computed with
// split-precision bf16 WMMA: x ~= hi + lo, x*y ~= hi*hi + hi*lo + lo*hi.
// 128x128 tile per workgroup, 8 waves of 16 rows, k-step 32.
// =====================================================================
#define GLDS 40   // padded LDS row stride (elems), 80B = multiple of 16B
__global__ __launch_bounds__(256) void gemm_split_bf16(
    const float* __restrict__ A, const float* __restrict__ Bw,
    float* __restrict__ C, int M, int N, int K) {
  __shared__ unsigned short a_hi[128 * GLDS], a_lo[128 * GLDS];
  __shared__ unsigned short b_hi[128 * GLDS], b_lo[128 * GLDS];
  const int tid  = threadIdx.x;
  const int wave = tid >> 5, lane = tid & 31;
  const int n0 = blockIdx.x * 128, m0 = blockIdx.y * 128;

  f32x8 acc[8];
#pragma unroll
  for (int i = 0; i < 8; ++i)
#pragma unroll
    for (int r = 0; r < 8; ++r) acc[i][r] = 0.0f;

  for (int k0 = 0; k0 < K; k0 += 32) {
    // ---- stage A and B tiles (f32 -> bf16 hi/lo) ----
#pragma unroll
    for (int v = 0; v < 4; ++v) {
      const int vec = tid + v * 256;      // 1024 float4 per 128x32 tile
      const int row = vec >> 3;           // 8 float4 per row
      const int c4  = (vec & 7) << 2;
      {
        float4 f = *(const float4*)(A + (size_t)(m0 + row) * K + k0 + c4);
        const float vv[4] = {f.x, f.y, f.z, f.w};
        const int off = row * GLDS + c4;
#pragma unroll
        for (int e = 0; e < 4; ++e) {
          unsigned short h = f2bf(vv[e]);
          a_hi[off + e] = h;
          a_lo[off + e] = f2bf(vv[e] - bf2f(h));
        }
      }
      {
        float4 f = *(const float4*)(Bw + (size_t)(n0 + row) * K + k0 + c4);
        const float vv[4] = {f.x, f.y, f.z, f.w};
        const int off = row * GLDS + c4;
#pragma unroll
        for (int e = 0; e < 4; ++e) {
          unsigned short h = f2bf(vv[e]);
          b_hi[off + e] = h;
          b_lo[off + e] = f2bf(vv[e] - bf2f(h));
        }
      }
    }
    __syncthreads();

    const bf16x16 ah = load_a_frag(a_hi + wave * 16 * GLDS, GLDS);
    const bf16x16 al = load_a_frag(a_lo + wave * 16 * GLDS, GLDS);
#pragma unroll
    for (int nt = 0; nt < 8; ++nt) {
      const bf16x16 bh = load_b_frag(b_hi + nt * 16 * GLDS, GLDS);
      const bf16x16 bl = load_b_frag(b_lo + nt * 16 * GLDS, GLDS);
      acc[nt] = wmma_bf16(ah, bh, acc[nt]);
      acc[nt] = wmma_bf16(ah, bl, acc[nt]);
      acc[nt] = wmma_bf16(al, bh, acc[nt]);
    }
    __syncthreads();
  }

  // C/D layout: row = r + 8*(lane>>4), col = lane&15
  const int rowb = m0 + wave * 16 + ((lane >> 4) << 3);
  const int col  = n0 + (lane & 15);
#pragma unroll
  for (int nt = 0; nt < 8; ++nt)
#pragma unroll
    for (int r = 0; r < 8; ++r)
      C[(size_t)(rowb + r) * N + col + nt * 16] = acc[nt][r];
}

// =====================================================================
// RoPE + split qkv (f32) into per-head bf16 Q/K/V buffers.
// qb: [NH][T][128], kb/vb: [NQG][T][128]
// =====================================================================
__global__ __launch_bounds__(256) void rope_split(
    const float* __restrict__ qkv, const float* __restrict__ cosb,
    const float* __restrict__ sinb, unsigned short* __restrict__ qb,
    unsigned short* __restrict__ kb, unsigned short* __restrict__ vb) {
  const int idx = blockIdx.x * 256 + threadIdx.x;   // T*QKV_N threads exactly
  const int t = idx / QKV_N, n = idx - t * QKV_N;
  const float val = qkv[idx];
  const int d = n & 127;
  if (n < NH * HEADD) {
    const int h = n >> 7;
    const float rot = (d < 64) ? -qkv[idx + 64] : qkv[idx - 64];
    const float o = val * cosb[t * HEADD + d] + rot * sinb[t * HEADD + d];
    qb[((size_t)h * TT + t) * HEADD + d] = f2bf(o);
  } else if (n < NH * HEADD + NQG * HEADD) {
    const int g = (n - NH * HEADD) >> 7;
    const float rot = (d < 64) ? -qkv[idx + 64] : qkv[idx - 64];
    const float o = val * cosb[t * HEADD + d] + rot * sinb[t * HEADD + d];
    kb[((size_t)g * TT + t) * HEADD + d] = f2bf(o);
  } else {
    const int g = (n - NH * HEADD - NQG * HEADD) >> 7;
    vb[((size_t)g * TT + t) * HEADD + d] = f2bf(val);
  }
}

// =====================================================================
// Flash attention with sliding window + tanh soft-cap.
// Workgroup = (q-tile of 128 rows, head). 8 waves x 16 q-rows each.
// Key blocks of 64 staged in LDS; online softmax in f32.
// =====================================================================
#define KBLK 64
#define KSTR 136   // K tile row stride (elems): 272B, mult of 16B
#define VSTR 72    // Vt / P stride (elems): 144B, mult of 16B
__global__ __launch_bounds__(256) void attn_kernel(
    const unsigned short* __restrict__ qb, const unsigned short* __restrict__ kbuf,
    const unsigned short* __restrict__ vbuf, float* __restrict__ attn) {
  __shared__ unsigned short k_t[KBLK * KSTR];        // [j][d]
  __shared__ unsigned short v_t[HEADD * VSTR];       // [d][j] (transposed)
  __shared__ unsigned short p_st[8 * 16 * VSTR];     // per-wave P stage

  const int tid = threadIdx.x, wave = tid >> 5, lane = tid & 31;
  const int i0 = blockIdx.x * 128;
  const int h  = blockIdx.y;
  const int g  = h >> 2;                 // Q_PER_KV = 4
  const int qrow0 = i0 + wave * 16;
  const int col   = lane & 15;
  const int rowb  = qrow0 + ((lane >> 4) << 3);   // first C-layout row

  // Q fragments (A-operand), kept in registers for all key blocks
  bf16x16 qf[4];
  {
    const unsigned short* qp = qb + ((size_t)h * TT + qrow0) * HEADD;
#pragma unroll
    for (int ks = 0; ks < 4; ++ks) qf[ks] = load_a_frag(qp + ks * 32, HEADD);
  }

  f32x8 o[8];
  float m_[8], l_[8];
#pragma unroll
  for (int i = 0; i < 8; ++i) {
#pragma unroll
    for (int r = 0; r < 8; ++r) o[i][r] = 0.0f;
    m_[i] = -__builtin_inff();
    l_[i] = 0.0f;
  }
  const float scale = 0.08838834764831845f;   // 1/sqrt(128)

  int jstart = i0 - (SWIN - 128);             // multiple of 64
  if (jstart < 0) jstart = 0;
  const int jend = i0 + 128;

  for (int jb = jstart; jb < jend; jb += KBLK) {
    // ---- stage K [j][d] into LDS ----
#if HAVE_ASYNC_LDS
#pragma unroll
    for (int v = 0; v < 4; ++v) {
      const int vec = tid + v * 256;          // 1024 x 16B transfers
      const int j  = vec >> 4;
      const int d8 = (vec & 15) << 3;
      typedef __attribute__((address_space(1))) v4i* g4p_t;
      typedef __attribute__((address_space(3))) v4i* l4p_t;
      g4p_t gsrc = (g4p_t)(kbuf + ((size_t)g * TT + jb + j) * HEADD + d8);
      l4p_t ldst = (l4p_t)&k_t[j * KSTR + d8];
      __builtin_amdgcn_global_load_async_to_lds_b128(gsrc, ldst, 0, 0);
    }
#else
#pragma unroll
    for (int v = 0; v < 4; ++v) {
      const int vec = tid + v * 256;          // 1024 x 8 elems
      const int j  = vec >> 4;
      const int d8 = (vec & 15) << 3;
      *(bf16x8*)&k_t[j * KSTR + d8] =
          *(const bf16x8*)(kbuf + ((size_t)g * TT + jb + j) * HEADD + d8);
    }
#endif
    // ---- stage V transposed [d][j] (needs transpose -> VALU path) ----
#pragma unroll
    for (int e = 0; e < 32; ++e) {
      const int idx = tid + e * 256;          // 8192 scalar elems
      const int j = idx >> 7, d = idx & 127;
      v_t[d * VSTR + j] = vbuf[((size_t)g * TT + jb + j) * HEADD + d];
    }
    // ---- prefetch next key block while this one is consumed ----
    if (jb + KBLK < jend && tid < KBLK) {
      const size_t nrow = (size_t)g * TT + jb + KBLK + tid;
      __builtin_prefetch(kbuf + nrow * HEADD, 0, 0);
      __builtin_prefetch(vbuf + nrow * HEADD, 0, 0);
    }
#if HAVE_ASYNC_LDS
    __builtin_amdgcn_s_wait_asynccnt(0);
#endif
    __syncthreads();

    // ---- S = Q K^T for 16 x 64 block ----
    f32x8 s[4];
#pragma unroll
    for (int nt = 0; nt < 4; ++nt) {
      f32x8 z;
#pragma unroll
      for (int r = 0; r < 8; ++r) z[r] = 0.0f;
#pragma unroll
      for (int ks = 0; ks < 4; ++ks) {
        const bf16x16 kf = load_b_frag(k_t + nt * 16 * KSTR + ks * 32, KSTR);
        z = wmma_bf16(qf[ks], kf, z);
      }
      s[nt] = z;
    }

    // ---- scale, soft-cap, mask, online softmax ----
    float mnew[8];
#pragma unroll
    for (int r = 0; r < 8; ++r) mnew[r] = m_[r];
#pragma unroll
    for (int nt = 0; nt < 4; ++nt) {
      const int j = jb + nt * 16 + col;
#pragma unroll
      for (int r = 0; r < 8; ++r) {
        float sv = s[nt][r] * scale;
        sv = CAPV * tanhf(sv * (1.0f / CAPV));
        const int i = rowb + r;
        const bool ok = (j <= i) && (i - j < SWIN);
        sv = ok ? sv : -__builtin_inff();
        s[nt][r] = sv;
        mnew[r] = fmaxf(mnew[r], sv);
      }
    }
#pragma unroll
    for (int r = 0; r < 8; ++r) {               // row max across the 16 lanes
      float mv = mnew[r];
      mv = fmaxf(mv, __shfl_xor(mv, 1, 32));
      mv = fmaxf(mv, __shfl_xor(mv, 2, 32));
      mv = fmaxf(mv, __shfl_xor(mv, 4, 32));
      mv = fmaxf(mv, __shfl_xor(mv, 8, 32));
      mnew[r] = mv;
    }
    float alpha[8], psum[8];
#pragma unroll
    for (int r = 0; r < 8; ++r) {
      alpha[r] = __expf(m_[r] - mnew[r]);       // exp(-inf)=0 on first block
      m_[r] = mnew[r];
      psum[r] = 0.0f;
    }
#pragma unroll
    for (int nt = 0; nt < 4; ++nt)
#pragma unroll
      for (int r = 0; r < 8; ++r) {
        const float p = __expf(s[nt][r] - m_[r]);
        s[nt][r] = p;
        psum[r] += p;
      }
#pragma unroll
    for (int r = 0; r < 8; ++r) l_[r] = l_[r] * alpha[r] + psum[r];
#pragma unroll
    for (int dnt = 0; dnt < 8; ++dnt)
#pragma unroll
      for (int r = 0; r < 8; ++r) o[dnt][r] *= alpha[r];

    // ---- P (C layout) -> LDS -> A layout ----
    unsigned short* pw = p_st + wave * 16 * VSTR;
#pragma unroll
    for (int nt = 0; nt < 4; ++nt)
#pragma unroll
      for (int r = 0; r < 8; ++r)
        pw[(r + ((lane >> 4) << 3)) * VSTR + nt * 16 + col] = f2bf(s[nt][r]);
    // same-wave LDS ops are in-order; compiler inserts s_wait_dscnt for reads

    // ---- O += P V ----
#pragma unroll
    for (int ks = 0; ks < 2; ++ks) {
      const bf16x16 pf = load_a_frag(pw + ks * 32, VSTR);
#pragma unroll
      for (int dnt = 0; dnt < 8; ++dnt) {
        const bf16x16 vf = load_b_frag(v_t + dnt * 16 * VSTR + ks * 32, VSTR);
        o[dnt] = wmma_bf16(pf, vf, o[dnt]);
      }
    }
    __syncthreads();
  }

  // ---- finalize: divide by row sum, store [t][h*128+d] f32 ----
  float inv[8];
#pragma unroll
  for (int r = 0; r < 8; ++r) {
    float lv = l_[r];
    lv += __shfl_xor(lv, 1, 32);
    lv += __shfl_xor(lv, 2, 32);
    lv += __shfl_xor(lv, 4, 32);
    lv += __shfl_xor(lv, 8, 32);
    inv[r] = 1.0f / lv;
  }
#pragma unroll
  for (int dnt = 0; dnt < 8; ++dnt)
#pragma unroll
    for (int r = 0; r < 8; ++r)
      attn[(size_t)(rowb + r) * EMBD + h * HEADD + dnt * 16 + col] =
          o[dnt][r] * inv[r];
}

// =====================================================================
extern "C" void kernel_launch(void* const* d_in, const int* in_sizes, int n_in,
                              void* d_out, int out_size, void* d_ws, size_t ws_size,
                              hipStream_t stream) {
  const float* x      = (const float*)d_in[0];
  const float* w_qkv  = (const float*)d_in[1];
  const float* w_proj = (const float*)d_in[2];
  const float* cosb   = (const float*)d_in[3];
  const float* sinb   = (const float*)d_in[4];
  float* out = (float*)d_out;

  char* ws = (char*)d_ws;
  float*          qkv  = (float*)(ws);                          // 25.2 MB
  float*          attn = (float*)(ws + (size_t)25165824);       // 16.8 MB
  unsigned short* qb   = (unsigned short*)(ws + (size_t)41943040); // 8.4 MB
  unsigned short* kb   = (unsigned short*)(ws + (size_t)50331648); // 2.1 MB
  unsigned short* vb   = (unsigned short*)(ws + (size_t)52428800); // 2.1 MB

  const dim3 blk(256);
  // 1) qkv = x @ w_qkv^T
  gemm_split_bf16<<<dim3(QKV_N / 128, TT / 128), blk, 0, stream>>>(
      x, w_qkv, qkv, TT, QKV_N, EMBD);
  // 2) RoPE + split to bf16 per-head buffers
  rope_split<<<(TT * QKV_N) / 256, blk, 0, stream>>>(qkv, cosb, sinb, qb, kb, vb);
  // 3) sliding-window flash attention
  attn_kernel<<<dim3(TT / 128, NH), blk, 0, stream>>>(qb, kb, vb, attn);
  // 4) out = attn @ w_proj^T
  gemm_split_bf16<<<dim3(EMBD / 128, TT / 128), blk, 0, stream>>>(
      attn, w_proj, out, TT, EMBD, EMBD);
}
